// ConvLRUModel_60911226192326
// MI455X (gfx1250) — matrix operand order repaired and checked by reading, not compiled, not run on hardware
//
#include <hip/hip_runtime.h>
#include <hip/hip_bf16.h>

typedef __attribute__((ext_vector_type(16))) _Float16 v16h;
typedef __attribute__((ext_vector_type(8)))  float    v8f;

#define WMMA_F16(A,B,C) __builtin_amdgcn_wmma_f32_16x16x32_f16(false,(A),false,(B),(short)0,(C),false,false)

static constexpr int Bsz = 2, Lsz = 16, Csz = 64, Hsz = 32, Wsz = 32, Fsz = 128, NHsz = 2, NBsz = 2;
static constexpr int PLANE  = Hsz * Wsz;                 // 1024
static constexpr int XELEMS = Bsz * Lsz * Csz * PLANE;   // 2,097,152
static constexpr int NIMG   = Bsz * Lsz;                 // 32

// ---------------------------------------------------------------------------
// 2D FFT (32x32) per (b,l,c) plane. Row pass = radix-2 across wave32 lanes
// (line length == wave size), column pass via LDS transpose.
// ---------------------------------------------------------------------------
__device__ __forceinline__ int brev5(int x) {
  return ((x & 1) << 4) | ((x & 2) << 2) | (x & 4) | ((x & 8) >> 2) | ((x & 16) >> 4);
}

__device__ __forceinline__ void fft32(float& re, float& im, int lane, float sign) {
  #pragma unroll
  for (int s = 0; s < 5; ++s) {
    int mh = 1 << s;
    float ang = sign * 6.28318530717958647692f * (float)(lane & (mh - 1)) / (float)(mh << 1);
    float tre = cosf(ang), tim = sinf(ang);
    float pre = __shfl_xor(re, mh, 32);
    float pim = __shfl_xor(im, mh, 32);
    bool  up  = (lane & mh) != 0;
    float ure = up ? pre : re,  uim = up ? pim : im;
    float vre = up ? re  : pre, vim = up ? im  : pim;
    float tvre = tre * vre - tim * vim;
    float tvim = tre * vim + tim * vre;
    re = up ? (ure - tvre) : (ure + tvre);
    im = up ? (uim - tvim) : (uim + tvim);
  }
}

template<bool INVERSE, bool REAL_IN>
__global__ __launch_bounds__(1024) void fft2_kernel(const float* __restrict__ sre,
                                                    const float* __restrict__ sim,
                                                    float* __restrict__ dre,
                                                    float* __restrict__ dimg) {
  __shared__ float lre[32][33];
  __shared__ float lim[32][33];
  const int tx = threadIdx.x, ty = threadIdx.y;
  const size_t base = (size_t)blockIdx.x * PLANE;
  const float sign = INVERSE ? 1.0f : -1.0f;

  int sx = brev5(tx);
  float re = sre[base + ty * 32 + sx];
  float im = REAL_IN ? 0.0f : sim[base + ty * 32 + sx];
  fft32(re, im, tx, sign);            // along W
  lre[ty][tx] = re; lim[ty][tx] = im;
  __syncthreads();
  int sy = brev5(tx);
  re = lre[sy][ty]; im = lim[sy][ty]; // wave ty now holds column ty, bit-reversed
  fft32(re, im, tx, sign);            // along H
  if (INVERSE) { re *= (1.0f / 1024.0f); im *= (1.0f / 1024.0f); }
  dre[base + tx * 32 + ty] = re;
  dimg[base + tx * 32 + ty] = im;
}

// ---------------------------------------------------------------------------
// Complex channel mix: out[bl,o,p] = sum_c A[bl,c,p] * W[o,c] (+ bias)(* ga)
// Implicit GEMM: M = 128 positions/tile, N = K = 64.  4 real WMMAs / complex.
// ---------------------------------------------------------------------------
template<bool APPLY_GA, bool REAL_ONLY>
__global__ __launch_bounds__(256) void cmix_kernel(const float* __restrict__ are,
                                                   const float* __restrict__ aim,
                                                   const float* __restrict__ wre,
                                                   const float* __restrict__ wim,
                                                   const float* __restrict__ br,
                                                   const float* __restrict__ bi,
                                                   const float* __restrict__ pl,
                                                   float* __restrict__ ore,
                                                   float* __restrict__ oim) {
  __shared__ _Float16 Ar[128][64];
  __shared__ _Float16 Ai[128][64];
  __shared__ _Float16 Wr[64][64];
  __shared__ _Float16 Wi[64][64];
  const int tid = threadIdx.x;
  const int bl  = blockIdx.x >> 3;
  const int pb  = (blockIdx.x & 7) << 7;
  const size_t plane = (size_t)bl * (Csz * PLANE);

  for (int idx = tid; idx < 128 * 64; idx += 256) {
    int k = idx >> 7, p = idx & 127;
    size_t g = plane + (size_t)k * PLANE + pb + p;
    Ar[p][k] = (_Float16)are[g];
    Ai[p][k] = (_Float16)aim[g];
  }
  for (int idx = tid; idx < 64 * 64; idx += 256) {
    int o = idx >> 6, k = idx & 63;
    Wr[o][k] = (_Float16)wre[o * 64 + k];
    Wi[o][k] = (_Float16)wim[o * 64 + k];
  }
  __syncthreads();

  const int wave = tid >> 5, lane = tid & 31;
  const int hi = lane >> 4, ln = lane & 15;
  const int my_n = wave & 3;
  const int mrow = wave >> 2;
  const int o = my_n * 16 + ln;

  v8f accre[4] = {};
  v8f accim[4] = {};

  #pragma unroll
  for (int kb = 0; kb < 2; ++kb) {
    v16h bfr, bfi;
    #pragma unroll
    for (int e = 0; e < 16; ++e) {
      int k = kb * 32 + (e < 8 ? 8 * hi + e : 16 + 8 * hi + (e - 8));
      bfr[e] = Wr[o][k];
      bfi[e] = Wi[o][k];
    }
    #pragma unroll
    for (int mi = 0; mi < 4; ++mi) {
      int p = (mrow * 4 + mi) * 16 + ln;
      v16h afr, afi, afin;
      #pragma unroll
      for (int e = 0; e < 8; ++e) {
        int k0 = kb * 32 + 8 * hi + e;
        int k1 = kb * 32 + 16 + 8 * hi + e;
        afr[e]     = Ar[p][k0]; afr[8 + e] = Ar[p][k1];
        afi[e]     = Ai[p][k0]; afi[8 + e] = Ai[p][k1];
      }
      #pragma unroll
      for (int e = 0; e < 16; ++e) afin[e] = -afi[e];
      accre[mi] = WMMA_F16(afr,  bfr, accre[mi]);
      accre[mi] = WMMA_F16(afin, bfi, accre[mi]);   // re -= Aim*Wim
      if (!REAL_ONLY) {
        accim[mi] = WMMA_F16(afr, bfi, accim[mi]);
        accim[mi] = WMMA_F16(afi, bfr, accim[mi]);
      }
    }
  }

  float bvr = br[o];
  float bvi = 0.0f;
  if (!REAL_ONLY) bvi = bi[o];
  #pragma unroll
  for (int mi = 0; mi < 4; ++mi) {
    #pragma unroll
    for (int r = 0; r < 8; ++r) {
      int pip = pb + (mrow * 4 + mi) * 16 + r + 8 * hi;
      size_t oaddr = plane + (size_t)o * PLANE + pip;
      float vr = accre[mi][r] + bvr;
      if (APPLY_GA) {
        int hh = pip >> 5;
        float ga = expf(pl[(2 * Csz + o) * Hsz + hh]);
        ore[oaddr] = vr * ga;
        oim[oaddr] = (accim[mi][r] + bvi) * ga;
      } else if (!REAL_ONLY) {
        ore[oaddr] = vr;
        oim[oaddr] = accim[mi][r] + bvi;
      } else {
        ore[oaddr] = vr;   // h.real only
      }
    }
  }
}

// ---------------------------------------------------------------------------
// LRU scan over L (sequential, per (b,c,h,w)); lambda recomputed in-register.
// ---------------------------------------------------------------------------
__global__ __launch_bounds__(256) void scan_kernel(float* __restrict__ hre,
                                                   float* __restrict__ him,
                                                   const float* __restrict__ pl) {
  int t   = blockIdx.x * 256 + threadIdx.x;  // 131072 total
  int b   = t >> 16;
  int rem = t & 65535;
  int c   = rem >> 10;
  int hw  = rem & 1023;
  int hh  = hw >> 5;
  float nu  = expf(pl[c * Hsz + hh]);
  float th  = expf(pl[(Csz + c) * Hsz + hh]);
  float r   = expf(-nu);
  float lre = r * cosf(th);
  float lim = r * sinf(th);
  float cre = 0.f, cim = 0.f;
  for (int l = 0; l < Lsz; ++l) {
    size_t a = ((size_t)(b * Lsz + l) * Csz + c) * PLANE + hw;
    float xr = hre[a], xi = him[a];
    float nre = lre * cre - lim * cim + xr;
    float nim = lre * cim + lim * cre + xi;
    cre = nre; cim = nim;
    hre[a] = cre; him[a] = cim;
  }
}

// ---------------------------------------------------------------------------
// LayerNorm over a group of S elements + residual add (out may alias res).
// ---------------------------------------------------------------------------
__global__ __launch_bounds__(256) void ln_residual_kernel(const float* __restrict__ t,
                                                          const float* __restrict__ w,
                                                          const float* __restrict__ bb,
                                                          const float* __restrict__ res,
                                                          float* __restrict__ out,
                                                          int S) {
  const int tid = threadIdx.x;
  const size_t gb = (size_t)blockIdx.x * S;
  float s = 0.f, ss = 0.f;
  for (int i = tid; i < S; i += 256) { float v = t[gb + i]; s += v; ss += v * v; }
  #pragma unroll
  for (int off = 16; off > 0; off >>= 1) {
    s  += __shfl_down(s,  off, 32);
    ss += __shfl_down(ss, off, 32);
  }
  __shared__ float rs[8], rss[8];
  int wave = tid >> 5, lane = tid & 31;
  if (lane == 0) { rs[wave] = s; rss[wave] = ss; }
  __syncthreads();
  if (tid == 0) {
    float a = 0.f, b2 = 0.f;
    for (int k = 0; k < 8; ++k) { a += rs[k]; b2 += rss[k]; }
    rs[0] = a; rss[0] = b2;
  }
  __syncthreads();
  float inv  = 1.0f / (float)S;
  float mean = rs[0] * inv;
  float var  = rss[0] * inv - mean * mean;
  float rstd = rsqrtf(var + 1e-5f);
  for (int i = tid; i < S; i += 256) {
    out[gb + i] = (t[gb + i] - mean) * rstd * w[i] + bb[i] + res[gb + i];
  }
}

// ---------------------------------------------------------------------------
// 3x3 SAME conv as implicit WMMA GEMM. One workgroup = one output row (M=32
// pixels) x COUT.  Patch + weight chunks staged in LDS as f16.
// ---------------------------------------------------------------------------
__device__ __forceinline__ float gelu_exact(float v) {
  return 0.5f * v * (1.0f + erff(v * 0.70710678118654752440f));
}

template<int CIN, int COUT, bool GELU>
__global__ __launch_bounds__(256) void conv3x3_kernel(const float* __restrict__ in,
                                                      const float* __restrict__ wgt,
                                                      const float* __restrict__ bias,
                                                      float* __restrict__ out) {
  __shared__ _Float16 patch[3][32][CIN];      // [ry][x][cin]
  __shared__ _Float16 wlds[9][COUT][32];      // [tap][cout][k-in-chunk]
  const int tid = threadIdx.x;
  const int img = blockIdx.x >> 5;
  const int y   = blockIdx.x & 31;

  for (int idx = tid; idx < 3 * 32 * CIN; idx += 256) {
    int x    = idx & 31;
    int rest = idx >> 5;
    int cin  = rest % CIN;
    int ry   = rest / CIN;
    int yy   = y + ry - 1;
    float v = (yy >= 0 && yy < 32) ? in[((size_t)(img * CIN + cin) * 32 + yy) * 32 + x] : 0.f;
    patch[ry][x][cin] = (_Float16)v;
  }

  const int wave = tid >> 5, lane = tid & 31;
  const int hi = lane >> 4, ln = lane & 15;
  constexpr int NCK = COUT / 16;
  constexpr int WPC = 8 / NCK;
  const int my_n = wave % NCK;
  const int mt0  = wave / NCK;
  const int co   = my_n * 16 + ln;

  v8f acc[2] = {};

  for (int cb = 0; cb < CIN / 32; ++cb) {
    __syncthreads();
    for (int idx = tid; idx < 9 * COUT * 32; idx += 256) {
      int kk   = idx & 31;
      int rest = idx >> 5;
      int c2   = rest % COUT;
      int tap  = rest / COUT;
      wlds[tap][c2][kk] = (_Float16)wgt[((size_t)(c2 * CIN) + cb * 32 + kk) * 9 + tap];
    }
    if (cb + 1 < CIN / 32 && tid < 64)   // hint next weight chunk into cache
      __builtin_prefetch(&wgt[((size_t)(tid * (COUT / 64) * CIN) + (cb + 1) * 32) * 9], 0, 1);
    __syncthreads();

    #pragma unroll
    for (int tap = 0; tap < 9; ++tap) {
      int ky = tap / 3, kx = tap % 3;
      v16h bf;
      #pragma unroll
      for (int e = 0; e < 16; ++e) bf[e] = wlds[tap][co][e < 8 ? 8 * hi + e : 16 + 8 * hi + (e - 8)];
      for (int mt = mt0; mt < 2; mt += WPC) {
        int px = mt * 16 + ln + kx - 1;
        v16h af = {};
        if (px >= 0 && px < 32) {
          #pragma unroll
          for (int e = 0; e < 8; ++e) {
            af[e]     = patch[ky][px][cb * 32 + 8 * hi + e];
            af[8 + e] = patch[ky][px][cb * 32 + 16 + 8 * hi + e];
          }
        }
        acc[mt] = WMMA_F16(af, bf, acc[mt]);
      }
    }
  }

  float bv = bias[co];
  for (int mt = mt0; mt < 2; mt += WPC) {
    #pragma unroll
    for (int r = 0; r < 8; ++r) {
      int px = mt * 16 + r + 8 * hi;
      float v = acc[mt][r] + bv;
      if (GELU) v = gelu_exact(v);
      out[((size_t)(img * COUT + co) * 32 + y) * 32 + px] = v;
    }
  }
}

// ---------------------------------------------------------------------------
// Orchestration
// ---------------------------------------------------------------------------
extern "C" void kernel_launch(void* const* d_in, const int* in_sizes, int n_in,
                              void* d_out, int out_size, void* d_ws, size_t ws_size,
                              hipStream_t stream) {
  (void)in_sizes; (void)n_in; (void)out_size; (void)ws_size;
  const float* x     = (const float*)d_in[0];
  const float* plog  = (const float*)d_in[1];
  const float* pBwr  = (const float*)d_in[2];
  const float* pBwi  = (const float*)d_in[3];
  const float* pBbr  = (const float*)d_in[4];
  const float* pBbi  = (const float*)d_in[5];
  const float* pCwr  = (const float*)d_in[6];
  const float* pCwi  = (const float*)d_in[7];
  const float* pCbr  = (const float*)d_in[8];
  const float* cinw  = (const float*)d_in[12];
  const float* cinb  = (const float*)d_in[13];
  const float* chw   = (const float*)d_in[14];
  const float* chb   = (const float*)d_in[15];
  const float* chlnw = (const float*)d_in[16];
  const float* chlnb = (const float*)d_in[17];
  const float* coutw = (const float*)d_in[18];
  const float* coutb = (const float*)d_in[19];
  const float* ln1w  = (const float*)d_in[10];
  const float* ln1b  = (const float*)d_in[11];
  const float* ln2w  = (const float*)d_in[20];
  const float* ln2b  = (const float*)d_in[21];

  float* ws   = (float*)d_ws;
  float* xcur = ws;                       // 2M floats
  float* aRe  = xcur + XELEMS;            // 2M
  float* aIm  = aRe  + XELEMS;            // 2M
  float* bRe  = aIm  + XELEMS;            // 2M (also h.real / ln1 input)
  float* bIm  = bRe  + XELEMS;            // 2M (also conv_out result)
  float* act0 = bIm  + XELEMS;            // 4M (y)
  float* act1 = act0 + (size_t)NIMG * Fsz * PLANE; // 4M (y2)

  hipMemcpyAsync(xcur, x, sizeof(float) * XELEMS, hipMemcpyDeviceToDevice, stream);

  for (int i = 0; i < NBsz; ++i) {
    const float* pl = plog + (size_t)i * 3 * Csz * Hsz;

    // FFT2(x) -> aRe/aIm
    fft2_kernel<false, true><<<Bsz * Lsz * Csz, dim3(32, 32), 0, stream>>>(xcur, nullptr, aRe, aIm);
    // B mix + ga -> bRe/bIm
    cmix_kernel<true, false><<<256, 256, 0, stream>>>(aRe, aIm,
        pBwr + (size_t)i * Csz * Csz, pBwi + (size_t)i * Csz * Csz,
        pBbr + i * Csz, pBbi + i * Csz, pl, bRe, bIm);
    // LRU scan over L (in place)
    scan_kernel<<<512, 256, 0, stream>>>(bRe, bIm, pl);
    // IFFT2 -> aRe/aIm
    fft2_kernel<true, false><<<Bsz * Lsz * Csz, dim3(32, 32), 0, stream>>>(bRe, bIm, aRe, aIm);
    // C mix (real part) -> bRe
    cmix_kernel<false, true><<<256, 256, 0, stream>>>(aRe, aIm,
        pCwr + (size_t)i * Csz * Csz, pCwi + (size_t)i * Csz * Csz,
        pCbr + i * Csz, nullptr, nullptr, bRe, nullptr);
    // ln1 + residual -> xcur
    ln_residual_kernel<<<NIMG, 256, 0, stream>>>(bRe,
        ln1w + (size_t)i * Csz * PLANE, ln1b + (size_t)i * Csz * PLANE,
        xcur, xcur, Csz * PLANE);
    // conv_in + gelu -> act0
    conv3x3_kernel<64, 128, true><<<NIMG * 32, 256, 0, stream>>>(xcur,
        cinw + (size_t)i * Fsz * Csz * 9, cinb + i * Fsz, act0);
    // hidden blocks
    for (int j = 0; j < NHsz; ++j) {
      size_t wo = (size_t)i * NHsz + j;
      conv3x3_kernel<128, 128, true><<<NIMG * 32, 256, 0, stream>>>(act0,
          chw + wo * Fsz * Fsz * 9, chb + wo * Fsz, act1);
      ln_residual_kernel<<<NIMG, 256, 0, stream>>>(act1,
          chlnw + wo * Fsz * PLANE, chlnb + wo * Fsz * PLANE,
          act0, act0, Fsz * PLANE);
    }
    // conv_out -> bIm (reused as scratch)
    conv3x3_kernel<128, 64, false><<<NIMG * 32, 256, 0, stream>>>(act0,
        coutw + (size_t)i * Csz * Fsz * 9, coutb + i * Csz, bIm);
    // ln2 + residual -> xcur
    ln_residual_kernel<<<NIMG, 256, 0, stream>>>(bIm,
        ln2w + (size_t)i * Csz * PLANE, ln2b + (size_t)i * Csz * PLANE,
        xcur, xcur, Csz * PLANE);
  }

  hipMemcpyAsync(d_out, xcur, sizeof(float) * XELEMS, hipMemcpyDeviceToDevice, stream);
}
